// RotarySelfAttention_42039139894212
// MI455X (gfx1250) — compile-verified, matrix operand order checked by reading, not run on hardware
//
#include <hip/hip_runtime.h>
#include <hip/hip_bf16.h>

#define DEV __device__ __forceinline__

typedef __attribute__((ext_vector_type(16))) __bf16        v16bf;
typedef __attribute__((ext_vector_type(8)))  float         v8f;
typedef __attribute__((ext_vector_type(4)))  unsigned int  u32x4;
typedef __attribute__((ext_vector_type(8)))  int           i32x8;
typedef __attribute__((ext_vector_type(4)))  int           i32x4;

#if defined(__has_builtin)
#  if __has_builtin(__builtin_amdgcn_tensor_load_to_lds) && \
      __has_builtin(__builtin_amdgcn_s_wait_tensorcnt)
#    define HAVE_TDM 1
#  else
#    define HAVE_TDM 0
#  endif
#else
#  define HAVE_TDM 0
#endif

constexpr int D_MODEL = 1024;
constexpr int N_HEADS = 16;
constexpr int D_HEAD  = 64;
constexpr int BATCH   = 4;
constexpr int SEQ     = 2048;
constexpr int MROWS   = BATCH * SEQ;   // 8192

// ---------- bf16 helpers (bit-exact round-to-nearest-even) ----------
DEV __bf16 tobf(float f) {
  union { float f; unsigned u; } v; v.f = f;
  unsigned r = v.u + 0x7FFFu + ((v.u >> 16) & 1u);
  unsigned short h = (unsigned short)(r >> 16);
  __bf16 b; __builtin_memcpy(&b, &h, 2); return b;
}
DEV float frombf(__bf16 b) {
  unsigned short h; __builtin_memcpy(&h, &b, 2);
  union { float f; unsigned u; } v; v.u = ((unsigned)h) << 16; return v.f;
}
DEV v8f vzero8() { v8f z = {0.f,0.f,0.f,0.f,0.f,0.f,0.f,0.f}; return z; }

DEV float redmax16(float v) {
  v = fmaxf(v, __shfl_xor(v, 1, 32));
  v = fmaxf(v, __shfl_xor(v, 2, 32));
  v = fmaxf(v, __shfl_xor(v, 4, 32));
  v = fmaxf(v, __shfl_xor(v, 8, 32));
  return v;
}
DEV float redsum16(float v) {
  v += __shfl_xor(v, 1, 32);
  v += __shfl_xor(v, 2, 32);
  v += __shfl_xor(v, 4, 32);
  v += __shfl_xor(v, 8, 32);
  return v;
}

DEV v8f wmma_bf16(v16bf a, v16bf b, v8f c) {
  return __builtin_amdgcn_wmma_f32_16x16x32_bf16(false, a, false, b, (short)0, c,
                                                 false, false);
}

#if HAVE_TDM
// ---------------------------------------------------------------------------
// TDM: DMA a 2D tile of bf16 (data_size=2B) from global -> LDS, with HW row
// padding so LDS rows land on our padded strides. tensor_dim == tile_dim
// (tile always fully in-bounds). D# packing per CDNA5 ISA 08_async_tensor §8.
//   pad_interval_code: 3 -> every 16 DWORDs (32 bf16 row), 4 -> every 32 DWORDs
//   pad_amount_code:   7 -> 8 DWORDs (16 bf16 pad)
// This toolchain exposes the 6-arg builtin:
//   (u32x4 g0, i32x8 g1, i32x4 g2, i32x4 g3, i32x8 extra, i32 cpol)
// ---------------------------------------------------------------------------
DEV void tdm_load_2d_bf16(const __bf16* gsrc, unsigned lds_byte_off,
                          unsigned d0, unsigned d1, unsigned stride0_elems,
                          unsigned pad_interval_code, unsigned pad_amount_code)
{
  const unsigned long long ga = (unsigned long long)(const char*)gsrc;
  u32x4 g0;
  g0[0] = 1u;                                            // count=1, user mode
  g0[1] = lds_byte_off;                                  // lds_addr
  g0[2] = (unsigned)(ga & 0xFFFFFFFFull);                // global_addr[31:0]
  g0[3] = (unsigned)((ga >> 32) & 0x01FFFFFFull)
        | (2u << 30);                                    // type=2 ("image")
  i32x8 g1;
  g1[0] = (int)((1u << 16)                               // data_size = 2 bytes
              | (1u << 20)                               // pad_enable
              | (pad_interval_code << 22)
              | (pad_amount_code << 25));
  g1[1] = (int)((d0 & 0xFFFFu) << 16);                   // tensor_dim0[15:0]
  g1[2] = (int)(((d0 >> 16) & 0xFFFFu) | ((d1 & 0xFFFFu) << 16)); // d0 hi|d1 lo
  g1[3] = (int)(((d1 >> 16) & 0xFFFFu) | ((d0 & 0xFFFFu) << 16)); // d1 hi|tile0
  g1[4] = (int)(d1 & 0xFFFFu);                           // tile_dim1, tile_dim2=0
  g1[5] = (int)stride0_elems;                            // dim0_stride[31:0]
  g1[6] = 0;                                             // stride0 hi | stride1 lo
  g1[7] = 0;
  const i32x4 gz4 = {0, 0, 0, 0};                        // 2D: groups 2/3 unused
  const i32x8 gz8 = {0, 0, 0, 0, 0, 0, 0, 0};
  __builtin_amdgcn_tensor_load_to_lds(g0, g1, gz4, gz4, gz8, 0);
}
DEV unsigned lds_off32(const void* p) { return (unsigned)(size_t)p; }
#endif

// ---------------------------------------------------------------------------
// Kernel 1: f32 -> bf16 cast (grid-stride)
// ---------------------------------------------------------------------------
__global__ void cast_bf16_kernel(const float* __restrict__ in,
                                 __bf16* __restrict__ out, int n) {
  int i = blockIdx.x * blockDim.x + threadIdx.x;
  const int stride = gridDim.x * blockDim.x;
  for (; i < n; i += stride) out[i] = tobf(in[i]);
}

// ---------------------------------------------------------------------------
// Kernel 2: bf16 GEMM  O[m,e] = sum_d A[m,d]*W[e,d] + bias[e]
//   mode 0: bf16 output [b,h,s,d] (QKV)   mode 1: f32 row-major (final proj)
// 256 thr / 8 waves, tile 128x64, K-step 32, wave owns 32x32 of C.
// TDM double-buffered LDS staging when available.
// ---------------------------------------------------------------------------
__global__ __launch_bounds__(256)
void gemm_bf16_kernel(const __bf16* __restrict__ A, const __bf16* __restrict__ W,
                      const float* __restrict__ bias, void* __restrict__ outp,
                      int mode)
{
#if HAVE_TDM
  __shared__ __bf16 As[2][128][48];
  __shared__ __bf16 Bs[2][64][48];
#else
  __shared__ __bf16 As1[128][48];
  __shared__ __bf16 Bs1[64][48];
#endif
  const int tid  = threadIdx.x;
  const int lane = tid & 31;
  const int wave = tid >> 5;
  const int wm   = wave >> 1;
  const int wn   = wave & 1;
  const int gm0  = blockIdx.x * 128;
  const int gn0  = blockIdx.y * 64;

  v8f acc[2][2];
  for (int i = 0; i < 2; ++i)
    for (int j = 0; j < 2; ++j) acc[i][j] = vzero8();

#if HAVE_TDM
  if (wave == 0) {   // prologue DMA for k0 = 0
    tdm_load_2d_bf16(A + (size_t)gm0 * D_MODEL, lds_off32(&As[0][0][0]),
                     32, 128, D_MODEL, 3, 7);
    tdm_load_2d_bf16(W + (size_t)gn0 * D_MODEL, lds_off32(&Bs[0][0][0]),
                     32, 64, D_MODEL, 3, 7);
    __builtin_amdgcn_s_wait_tensorcnt(0);
  }
  __syncthreads();

  for (int k0 = 0; k0 < D_MODEL; k0 += 32) {
    const int cur = (k0 >> 5) & 1;
    const int nxt = cur ^ 1;
    if (wave == 0 && (k0 + 32) < D_MODEL) {   // async prefetch of next K-slab
      tdm_load_2d_bf16(A + (size_t)gm0 * D_MODEL + k0 + 32,
                       lds_off32(&As[nxt][0][0]), 32, 128, D_MODEL, 3, 7);
      tdm_load_2d_bf16(W + (size_t)gn0 * D_MODEL + k0 + 32,
                       lds_off32(&Bs[nxt][0][0]), 32, 64, D_MODEL, 3, 7);
    }
    const int kh = (lane >> 4) * 16;
    v16bf af[2], bfr[2];
    for (int mt = 0; mt < 2; ++mt)
      af[mt]  = *(const v16bf*)&As[cur][wm * 32 + mt * 16 + (lane & 15)][kh];
    for (int nt = 0; nt < 2; ++nt)
      bfr[nt] = *(const v16bf*)&Bs[cur][wn * 32 + nt * 16 + (lane & 15)][kh];
    for (int mt = 0; mt < 2; ++mt)
      for (int nt = 0; nt < 2; ++nt)
        acc[mt][nt] = wmma_bf16(af[mt], bfr[nt], acc[mt][nt]);

    if (wave == 0) __builtin_amdgcn_s_wait_tensorcnt(0);
    __syncthreads();
  }
#else
  const int arow = tid >> 1, ahalf = tid & 1;
  const int brow = tid >> 2, bq    = tid & 3;
  for (int k0 = 0; k0 < D_MODEL; k0 += 32) {
    __syncthreads();
    *(v16bf*)&As1[arow][ahalf * 16] =
        *(const v16bf*)(A + (size_t)(gm0 + arow) * D_MODEL + k0 + ahalf * 16);
    *(u32x4*)&Bs1[brow][bq * 8] =
        *(const u32x4*)(W + (size_t)(gn0 + brow) * D_MODEL + k0 + bq * 8);
    __syncthreads();
    const int kh = (lane >> 4) * 16;
    v16bf af[2], bfr[2];
    for (int mt = 0; mt < 2; ++mt)
      af[mt]  = *(const v16bf*)&As1[wm * 32 + mt * 16 + (lane & 15)][kh];
    for (int nt = 0; nt < 2; ++nt)
      bfr[nt] = *(const v16bf*)&Bs1[wn * 32 + nt * 16 + (lane & 15)][kh];
    for (int mt = 0; mt < 2; ++mt)
      for (int nt = 0; nt < 2; ++nt)
        acc[mt][nt] = wmma_bf16(af[mt], bfr[nt], acc[mt][nt]);
  }
#endif

  for (int mt = 0; mt < 2; ++mt) {
    for (int nt = 0; nt < 2; ++nt) {
      const int gn = gn0 + wn * 32 + nt * 16 + (lane & 15);
      const float bv = bias[gn];
      for (int j = 0; j < 8; ++j) {
        const int m = gm0 + wm * 32 + mt * 16 + ((lane < 16) ? j : j + 8);
        const float val = acc[mt][nt][j] + bv;
        if (mode == 0) {
          __bf16* o = (__bf16*)outp;
          const int b = m >> 11, s = m & (SEQ - 1);
          const int h = gn >> 6, d = gn & 63;
          o[(((size_t)(b * N_HEADS + h) * SEQ + s) << 6) + d] = tobf(val);
        } else {
          float* o = (float*)outp;
          o[(size_t)m * D_MODEL + gn] = val;
        }
      }
    }
  }
}

// ---------------------------------------------------------------------------
// Kernel 3: in-place RoPE on bf16 [b,h,s,d]; one thread per (d, d+32) pair
// ---------------------------------------------------------------------------
__global__ void rope_kernel(__bf16* __restrict__ Q,
                            const float* __restrict__ cosb,
                            const float* __restrict__ sinb)
{
  const int idx = blockIdx.x * blockDim.x + threadIdx.x;
  if (idx >= BATCH * N_HEADS * SEQ * 32) return;
  const int d  = idx & 31;
  const int s  = (idx >> 5) & (SEQ - 1);
  const int bh = idx >> 16;
  const size_t base = (((size_t)bh * SEQ) + s) << 6;
  const float q1 = frombf(Q[base + d]);
  const float q2 = frombf(Q[base + d + 32]);
  const float c1 = cosb[s * 64 + d],      s1 = sinb[s * 64 + d];
  const float c2 = cosb[s * 64 + d + 32], s2 = sinb[s * 64 + d + 32];
  Q[base + d]      = tobf(q1 * c1 - q2 * s1);
  Q[base + d + 32] = tobf(q2 * c2 + q1 * s2);
}

// ---------------------------------------------------------------------------
// Kernel 4: flash attention per (b,h). 256 thr = 8 waves x 32 q-rows.
// K tile via TDM DMA (overlapped with manual transposed V staging).
// ---------------------------------------------------------------------------
__global__ __launch_bounds__(256)
void attn_kernel(const __bf16* __restrict__ Q, const __bf16* __restrict__ K,
                 const __bf16* __restrict__ V,
                 const unsigned char* __restrict__ mask,
                 __bf16* __restrict__ ctx)
{
  __shared__ __bf16 Ks[64][80];        // [key][d]   160 B stride (TDM-padded)
  __shared__ __bf16 Vs[64][80];        // [d][key]   transposed
  __shared__ __bf16 Ps[8][32][80];     // wave-private P staging

  const int tid   = threadIdx.x;
  const int lane  = tid & 31;
  const int wave  = tid >> 5;
  const int bh    = blockIdx.y;
  const int b     = bh >> 4;
  const int h     = bh & 15;
  const int qbase = blockIdx.x * 256 + wave * 32;

  v16bf qa[2][2];
  for (int mt = 0; mt < 2; ++mt)
    for (int kh = 0; kh < 2; ++kh) {
      const int row = qbase + mt * 16 + (lane & 15);
      qa[mt][kh] = *(const v16bf*)(Q + ((((size_t)bh * SEQ) + row) << 6) +
                                   kh * 32 + ((lane >> 4) * 16));
    }

  v8f co[2][4];
  for (int mt = 0; mt < 2; ++mt)
    for (int dt = 0; dt < 4; ++dt) co[mt][dt] = vzero8();
  float rm[2][8], rl[2][8];
  for (int mt = 0; mt < 2; ++mt)
    for (int j = 0; j < 8; ++j) { rm[mt][j] = -3.0e38f; rl[mt][j] = 0.f; }

  const int krow = tid >> 2, kpart = tid & 3;

  for (int kt = 0; kt < SEQ / 64; ++kt) {
    __syncthreads();
    const size_t tilebase = (((size_t)bh * SEQ) + kt * 64) << 6;
#if HAVE_TDM
    if (wave == 0)   // K tile: 64(d) x 64(keys), row stride 64, pad 32B/row
      tdm_load_2d_bf16(K + tilebase, lds_off32(&Ks[0][0]), 64, 64, 64, 4, 7);
    {
      const size_t vg = tilebase + ((size_t)krow << 6);
      v16bf vv = *(const v16bf*)(V + vg + kpart * 16);
      for (int i = 0; i < 16; ++i) Vs[kpart * 16 + i][krow] = vv[i];
    }
    if (wave == 0) __builtin_amdgcn_s_wait_tensorcnt(0);
#else
    {
      const size_t kg = tilebase + ((size_t)krow << 6);
      *(v16bf*)&Ks[krow][kpart * 16] = *(const v16bf*)(K + kg + kpart * 16);
      v16bf vv = *(const v16bf*)(V + kg + kpart * 16);
      for (int i = 0; i < 16; ++i) Vs[kpart * 16 + i][krow] = vv[i];
    }
#endif
    __syncthreads();

    // S = Q * K^T
    v8f sc[2][4];
    for (int nt = 0; nt < 4; ++nt) {
      v16bf kb0 = *(const v16bf*)&Ks[nt * 16 + (lane & 15)][(lane >> 4) * 16];
      v16bf kb1 = *(const v16bf*)&Ks[nt * 16 + (lane & 15)][32 + (lane >> 4) * 16];
      for (int mt = 0; mt < 2; ++mt) {
        v8f t = vzero8();
        t = wmma_bf16(qa[mt][0], kb0, t);
        t = wmma_bf16(qa[mt][1], kb1, t);
        sc[mt][nt] = t;
      }
    }

    // scale + key mask
    for (int nt = 0; nt < 4; ++nt) {
      const int kg = kt * 64 + nt * 16 + (lane & 15);
      const bool keep = mask[b * SEQ + kg] != 0;
      for (int mt = 0; mt < 2; ++mt)
        for (int j = 0; j < 8; ++j) {
          const float v = sc[mt][nt][j] * 0.125f;
          sc[mt][nt][j] = keep ? v : -3.0e38f;
        }
    }

    // online softmax
    for (int mt = 0; mt < 2; ++mt)
      for (int j = 0; j < 8; ++j) {
        float mx = fmaxf(fmaxf(sc[mt][0][j], sc[mt][1][j]),
                         fmaxf(sc[mt][2][j], sc[mt][3][j]));
        mx = redmax16(mx);
        const float nm   = fmaxf(rm[mt][j], mx);
        const float corr = __expf(rm[mt][j] - nm);
        rm[mt][j] = nm;
        float ssum = 0.f;
        for (int nt = 0; nt < 4; ++nt) {
          const float p = __expf(sc[mt][nt][j] - nm);
          sc[mt][nt][j] = p;
          ssum += p;
        }
        ssum = redsum16(ssum);
        rl[mt][j] = rl[mt][j] * corr + ssum;
        for (int dt = 0; dt < 4; ++dt) co[mt][dt][j] *= corr;
      }

    // P -> wave-private LDS (C-layout f32 -> A-layout bf16)
    for (int mt = 0; mt < 2; ++mt)
      for (int nt = 0; nt < 4; ++nt)
        for (int j = 0; j < 8; ++j) {
          const int ml = mt * 16 + ((lane < 16) ? j : j + 8);
          const int n  = nt * 16 + (lane & 15);
          Ps[wave][ml][n] = tobf(sc[mt][nt][j]);
        }
    asm volatile("s_wait_dscnt 0" ::: "memory");

    // O += P * V
    for (int kh = 0; kh < 2; ++kh) {
      v16bf pa[2];
      for (int mt = 0; mt < 2; ++mt)
        pa[mt] = *(const v16bf*)&Ps[wave][mt * 16 + (lane & 15)]
                                  [kh * 32 + (lane >> 4) * 16];
      for (int dt = 0; dt < 4; ++dt) {
        v16bf vb = *(const v16bf*)&Vs[dt * 16 + (lane & 15)]
                                     [kh * 32 + (lane >> 4) * 16];
        for (int mt = 0; mt < 2; ++mt)
          co[mt][dt] = wmma_bf16(pa[mt], vb, co[mt][dt]);
      }
    }
  }

  // finalize
  for (int mt = 0; mt < 2; ++mt)
    for (int j = 0; j < 8; ++j) {
      const float inv = 1.0f / rl[mt][j];
      const int ml = mt * 16 + ((lane < 16) ? j : j + 8);
      const int sg = qbase + ml;
      for (int dt = 0; dt < 4; ++dt) {
        const int d = dt * 16 + (lane & 15);
        ctx[((size_t)(b * SEQ + sg)) * D_MODEL + h * 64 + d] =
            tobf(co[mt][dt][j] * inv);
      }
    }
}

// ---------------------------------------------------------------------------
extern "C" void kernel_launch(void* const* d_in, const int* in_sizes, int n_in,
                              void* d_out, int out_size, void* d_ws, size_t ws_size,
                              hipStream_t stream) {
  (void)in_sizes; (void)n_in; (void)out_size; (void)ws_size;
  const float*         x    = (const float*)d_in[0];
  const unsigned char* mask = (const unsigned char*)d_in[1];
  const float*         cosb = (const float*)d_in[2];
  const float*         sinb = (const float*)d_in[3];
  const float*         Wq   = (const float*)d_in[4];
  const float*         bq   = (const float*)d_in[5];
  const float*         Wk   = (const float*)d_in[6];
  const float*         bk   = (const float*)d_in[7];
  const float*         Wv   = (const float*)d_in[8];
  const float*         bv   = (const float*)d_in[9];
  const float*         Wo   = (const float*)d_in[10];
  const float*         bo   = (const float*)d_in[11];
  float*               out  = (float*)d_out;

  char* ws = (char*)d_ws;
  const size_t MB = 1024 * 1024;
  __bf16* xb   = (__bf16*)(ws + 0);
  __bf16* Wqb  = (__bf16*)(ws + 16 * MB);
  __bf16* Wkb  = (__bf16*)(ws + 18 * MB);
  __bf16* Wvb  = (__bf16*)(ws + 20 * MB);
  __bf16* Wob  = (__bf16*)(ws + 22 * MB);
  __bf16* Qb   = (__bf16*)(ws + 24 * MB);
  __bf16* Kb   = (__bf16*)(ws + 40 * MB);
  __bf16* Vb   = (__bf16*)(ws + 56 * MB);
  __bf16* Ctxb = (__bf16*)(ws + 72 * MB);

  cast_bf16_kernel<<<2048, 256, 0, stream>>>(x,  xb,  MROWS * D_MODEL);
  cast_bf16_kernel<<<512,  256, 0, stream>>>(Wq, Wqb, D_MODEL * D_MODEL);
  cast_bf16_kernel<<<512,  256, 0, stream>>>(Wk, Wkb, D_MODEL * D_MODEL);
  cast_bf16_kernel<<<512,  256, 0, stream>>>(Wv, Wvb, D_MODEL * D_MODEL);
  cast_bf16_kernel<<<512,  256, 0, stream>>>(Wo, Wob, D_MODEL * D_MODEL);

  dim3 gg(MROWS / 128, D_MODEL / 64), gb(256);
  gemm_bf16_kernel<<<gg, gb, 0, stream>>>(xb, Wqb, bq, (void*)Qb, 0);
  gemm_bf16_kernel<<<gg, gb, 0, stream>>>(xb, Wkb, bk, (void*)Kb, 0);
  gemm_bf16_kernel<<<gg, gb, 0, stream>>>(xb, Wvb, bv, (void*)Vb, 0);

  const int nrope = BATCH * N_HEADS * SEQ * 32;
  rope_kernel<<<(nrope + 255) / 256, 256, 0, stream>>>(Qb, cosb, sinb);
  rope_kernel<<<(nrope + 255) / 256, 256, 0, stream>>>(Kb, cosb, sinb);

  attn_kernel<<<dim3(SEQ / 256, BATCH * N_HEADS), 256, 0, stream>>>(
      Qb, Kb, Vb, mask, Ctxb);

  gemm_bf16_kernel<<<gg, gb, 0, stream>>>(Ctxb, Wob, bo, (void*)out, 1);
}
